// attention_1_71734543778341
// MI455X (gfx1250) — matrix operand order repaired and checked by reading, not compiled, not run on hardware
//
#include <hip/hip_runtime.h>
#include <hip/hip_bf16.h>

typedef __attribute__((ext_vector_type(16))) __bf16          v16bf;
typedef __attribute__((ext_vector_type(8)))  float           v8f;
typedef __attribute__((ext_vector_type(16))) unsigned short  v16us;

#define H 256
#define B 64
#define D 2048

// ---------------------------------------------------------------- helpers
__device__ __forceinline__ unsigned short f2bf(float f) {
    // round-to-nearest-even fp32 -> bf16
    unsigned int u = __float_as_uint(f);
    unsigned int r = u + 0x7FFFu + ((u >> 16) & 1u);
    return (unsigned short)(r >> 16);
}

// Branch-free tanh on hardware transcendentals (CDNA5 has V_TANH_F32; else
// build from V_EXP_F32 + V_RCP_F32: tanh(x) = sign(x)*(1-e)/(1+e),
// e = exp2(-2*log2(e)*|x|) in (0,1] -> no overflow, no divergence).
__device__ __forceinline__ float fast_tanh(float x) {
#if __has_builtin(__builtin_amdgcn_tanhf)
    return __builtin_amdgcn_tanhf(x);
#else
    float ax = __builtin_fabsf(x);
    float e  = __builtin_amdgcn_exp2f(ax * -2.8853900817779268f);
    float r  = (1.0f - e) * __builtin_amdgcn_rcpf(1.0f + e);
    return __builtin_copysignf(r, x);
#endif
}

__device__ __forceinline__ float fast_exp(float x) {
    return __builtin_amdgcn_exp2f(x * 1.4426950408889634f);
}

struct U16x16 { uint4 a, b; };   // 32 bytes == v16bf

// ---------------------------------------------------------------- kernel 1: W1 fp32 -> bf16
__global__ __launch_bounds__(256) void k_cvt_w1(const float* __restrict__ W1,
                                                unsigned short* __restrict__ w1bf) {
    int i = blockIdx.x * 256 + threadIdx.x;   // grid 256 -> 65536 elems
    w1bf[i] = f2bf(W1[i]);
}

// ---------------------------------------------------------------- kernel 2: g23 = ctx@W2.T + b2 + hid@W3.T
__global__ __launch_bounds__(256) void k_g23(const float* __restrict__ ctx,
                                             const float* __restrict__ hid,
                                             const float* __restrict__ W2,
                                             const float* __restrict__ b2,
                                             const float* __restrict__ W3,
                                             float* __restrict__ g23) {
    int b = blockIdx.x, h = threadIdx.x;
    __shared__ float cs[H], hs[H];
    cs[h] = ctx[b * H + h];
    hs[h] = hid[b * H + h];
    __syncthreads();
    float s = b2[h];
    const float* w2 = W2 + h * H;
    const float* w3 = W3 + h * H;
    #pragma unroll 8
    for (int k = 0; k < H; ++k) s += cs[k] * w2[k] + hs[k] * w3[k];
    g23[b * H + h] = s;
}

// ---------------------------------------------------------------- kernel 3: fused g1 GEMM + tanh + W4 dot -> scores
// grid (D/128, B), block 256 (8 waves x 16 d-rows)
// LDS: W1 bf16, row stride padded to 264 bf16 (33 uint4) to spread banks.
__global__ __launch_bounds__(256) void k_scores(const float* __restrict__ question,
                                                const unsigned short* __restrict__ w1bf,
                                                const float* __restrict__ g23,
                                                const float* __restrict__ W4,
                                                float* __restrict__ scores) {
    extern __shared__ unsigned char smem[];
    uint4* w1q  = (uint4*)smem;                     // 256 rows * 33 uint4 = 135168 B
    float* g23s = (float*)(smem + 256 * 33 * 16);   // 256 floats
    float* w4s  = g23s + H;                         // 256 floats

    const int tid  = threadIdx.x;
    const int b    = blockIdx.y;
    const int dblk = blockIdx.x;

    // stage bf16 W1 into LDS (padded rows), plus g23 row and W4
    {
        const uint4* src = (const uint4*)w1bf;      // 256 rows * 32 uint4
        for (int i = tid; i < 256 * 32; i += 256) {
            int r = i >> 5, c = i & 31;
            w1q[r * 33 + c] = src[i];
        }
        g23s[tid] = g23[b * H + tid];
        w4s[tid]  = W4[tid];
    }
    __syncthreads();

    const int wave = tid >> 5;
    const int lane = tid & 31;
    const int m    = lane & 15;     // row within 16x16 tile (A) / col (B,C,D)
    const int kh   = lane >> 4;     // lane-half
    const int d0   = dblk * 128 + wave * 16;

    // ---- A operands: 16 d-rows x 256 k as 8 bf16 chunks, documented 16-bit A layout:
    //   elems 0..7  -> K = kc*32 + kh*8 + e
    //   elems 8..15 -> K = kc*32 + 16 + kh*8 + (e-8)
    v16bf aop[8];
    const float* qrow = question + (size_t)(b * D + d0 + m) * H;
    #pragma unroll
    for (int kc = 0; kc < 8; ++kc) {
        const float* p0 = qrow + kc * 32 + kh * 8;
        const float* p1 = qrow + kc * 32 + 16 + kh * 8;
        float4 f0 = *(const float4*)(p0);
        float4 f1 = *(const float4*)(p0 + 4);
        float4 f2 = *(const float4*)(p1);
        float4 f3 = *(const float4*)(p1 + 4);
        v16us t;
        t[0]  = f2bf(f0.x); t[1]  = f2bf(f0.y); t[2]  = f2bf(f0.z); t[3]  = f2bf(f0.w);
        t[4]  = f2bf(f1.x); t[5]  = f2bf(f1.y); t[6]  = f2bf(f1.z); t[7]  = f2bf(f1.w);
        t[8]  = f2bf(f2.x); t[9]  = f2bf(f2.y); t[10] = f2bf(f2.z); t[11] = f2bf(f2.w);
        t[12] = f2bf(f3.x); t[13] = f2bf(f3.y); t[14] = f2bf(f3.z); t[15] = f2bf(f3.w);
        aop[kc] = __builtin_bit_cast(v16bf, t);
    }

    float sacc[8] = {0.f, 0.f, 0.f, 0.f, 0.f, 0.f, 0.f, 0.f};

    for (int ht = 0; ht < 16; ++ht) {
        const int h0 = ht * 16;
        v8f acc = {};
        // B operand lane layout (16-bit, K=32): col n = lane&15, K = kh*8 + e (e<8),
        // 16 + kh*8 + (e-8) (e>=8) from W1 row (h0+n)  ->  two uint4 LDS loads.
        const int base = (h0 + m) * 33 + kh;
        #pragma unroll
        for (int kc = 0; kc < 8; ++kc) {
            U16x16 u;
            u.a = w1q[base + kc * 4];
            u.b = w1q[base + kc * 4 + 2];
            v16bf bop = __builtin_bit_cast(v16bf, u);
            acc = __builtin_amdgcn_wmma_f32_16x16x32_bf16(
                false, aop[kc], false, bop, (short)0, acc, false, false);
        }
        // epilogue: D elem r -> row d0 + r + 8*kh, col h0 + m
        const float g  = g23s[h0 + m];
        const float w4 = w4s[h0 + m];
        #pragma unroll
        for (int r = 0; r < 8; ++r) {
            float v = fast_tanh(acc[r] + g);
            sacc[r] += v * w4;
        }
    }

    // reduce over the 16 h-columns held by lanes n=0..15 of each half
    #pragma unroll
    for (int r = 0; r < 8; ++r) {
        float v = sacc[r];
        v += __shfl_xor(v, 1);
        v += __shfl_xor(v, 2);
        v += __shfl_xor(v, 4);
        v += __shfl_xor(v, 8);
        sacc[r] = v;
    }
    if (m == 0) {
        #pragma unroll
        for (int r = 0; r < 8; ++r)
            scores[b * D + d0 + kh * 8 + r] = sacc[r];
    }
}

// ---------------------------------------------------------------- kernel 4: softmax over D per batch
__global__ __launch_bounds__(256) void k_softmax(const float* __restrict__ scores,
                                                 float* __restrict__ attn) {
    const int b = blockIdx.x, tid = threadIdx.x;
    __shared__ float red[256];
    const float* s = scores + b * D;
    float v[8], e[8];
    float mx = -__builtin_inff();
    #pragma unroll
    for (int i = 0; i < 8; ++i) { v[i] = s[tid + i * 256]; mx = fmaxf(mx, v[i]); }
    red[tid] = mx; __syncthreads();
    for (int off = 128; off; off >>= 1) {
        if (tid < off) red[tid] = fmaxf(red[tid], red[tid + off]);
        __syncthreads();
    }
    mx = red[0]; __syncthreads();
    float sum = 0.f;
    #pragma unroll
    for (int i = 0; i < 8; ++i) { e[i] = fast_exp(v[i] - mx); sum += e[i]; }
    red[tid] = sum; __syncthreads();
    for (int off = 128; off; off >>= 1) {
        if (tid < off) red[tid] += red[tid + off];
        __syncthreads();
    }
    const float inv = 1.0f / red[0];
    #pragma unroll
    for (int i = 0; i < 8; ++i) attn[b * D + tid + i * 256] = e[i] * inv;
}

// ---------------------------------------------------------------- kernel 5: partial weighted sums (deterministic, no atomics)
// grid (8, B): chunk c covers 256 d-rows
__global__ __launch_bounds__(256) void k_wsum_part(const float* __restrict__ attn,
                                                   const float* __restrict__ question,
                                                   float* __restrict__ partial) {
    const int b = blockIdx.y, c = blockIdx.x, tid = threadIdx.x;
    __shared__ float a[256];
    a[tid] = attn[b * D + c * 256 + tid];
    __syncthreads();
    float acc = 0.f;
    const float* q = question + (size_t)(b * D + c * 256) * H + tid;
    #pragma unroll 8
    for (int j = 0; j < 256; ++j) acc += a[j] * q[(size_t)j * H];
    partial[(b * 8 + c) * H + tid] = acc;
}

// ---------------------------------------------------------------- kernel 6: finalize out[b,h] = sum_c partial
__global__ __launch_bounds__(256) void k_wsum_final(const float* __restrict__ partial,
                                                    float* __restrict__ out) {
    const int i = blockIdx.x * 256 + threadIdx.x;   // grid 64 -> 16384
    const int b = i >> 8, h = i & 255;
    float s = 0.f;
    #pragma unroll
    for (int c = 0; c < 8; ++c) s += partial[(b * 8 + c) * H + h];
    out[i] = s;
}

// ---------------------------------------------------------------- launch
extern "C" void kernel_launch(void* const* d_in, const int* in_sizes, int n_in,
                              void* d_out, int out_size, void* d_ws, size_t ws_size,
                              hipStream_t stream) {
    (void)in_sizes; (void)n_in; (void)out_size; (void)ws_size;
    const float* context  = (const float*)d_in[0];
    const float* question = (const float*)d_in[1];
    const float* hidden   = (const float*)d_in[2];
    const float* W1       = (const float*)d_in[3];
    const float* W2       = (const float*)d_in[4];
    const float* b2       = (const float*)d_in[5];
    const float* W3       = (const float*)d_in[6];
    const float* W4       = (const float*)d_in[7];
    // d_in[8] = b4: a per-batch constant added to every score -> softmax-invariant, skipped.
    float* out = (float*)d_out;

    char* ws = (char*)d_ws;
    unsigned short* w1bf  = (unsigned short*)(ws);            // 131072 B
    float* g23            = (float*)(ws + 131072);            //  65536 B
    float* scores         = (float*)(ws + 196608);            // 524288 B
    float* attn           = (float*)(ws + 720896);            // 524288 B
    float* partial        = (float*)(ws + 1245184);           // 524288 B

    k_cvt_w1<<<256, 256, 0, stream>>>(W1, w1bf);
    k_g23<<<B, 256, 0, stream>>>(context, hidden, W2, b2, W3, g23);

    const size_t lds_bytes = 256 * 33 * 16 + H * 4 + H * 4;   // 137216 B (of 320 KB/WGP)
    k_scores<<<dim3(D / 128, B), 256, lds_bytes, stream>>>(question, w1bf, g23, W4, scores);

    k_softmax<<<B, 256, 0, stream>>>(scores, attn);
    k_wsum_part<<<dim3(8, B), 256, 0, stream>>>(attn, question, partial);
    k_wsum_final<<<64, 256, 0, stream>>>(partial, out);
}
